// Attention_22771916604088
// MI455X (gfx1250) — matrix operand order repaired and checked by reading, not compiled, not run on hardware
//
#include <hip/hip_runtime.h>

// ---------------------------------------------------------------------------
// Problem constants (from reference): B=2, L=2048, H=2048, NH=16, DH=128
// ---------------------------------------------------------------------------
#define NB   2
#define LL   2048
#define HH   2048
#define NHH  16
#define DHH  128
#define MTOT (NB * LL)   // 4096 rows for the big GEMMs

typedef __attribute__((ext_vector_type(16))) __bf16 v16bf;
typedef __attribute__((ext_vector_type(8)))  float  v8f;
typedef int v4i_vs __attribute__((vector_size(16)));  // matches builtin param type

#define WMMA_BF16(A, B, C) \
  __builtin_amdgcn_wmma_f32_16x16x32_bf16(false, (A), false, (B), (short)0, (C), false, false)

// ---------------------------------------------------------------------------
// CDNA5 async global->LDS copies (ASYNCcnt path), guarded so the file still
// compiles (sync fallback) if the toolchain doesn't declare the builtins.
// Probe result: builtin exists on ROCm 7.2 with signature
//   (int vector_size(16)* src, int vector_size(16)* dst, Imm offset, Imm cpol)
// ---------------------------------------------------------------------------
#if defined(__has_builtin)
#if __has_builtin(__builtin_amdgcn_global_load_async_to_lds_b128) && \
    __has_builtin(__builtin_amdgcn_s_wait_asynccnt)
#define USE_ASYNC_LDS 1
#endif
#endif
#ifndef USE_ASYNC_LDS
#define USE_ASYNC_LDS 0
#endif

#if USE_ASYNC_LDS
__device__ __forceinline__ void async_cp_b128(const __bf16* gsrc, __bf16* ldst) {
  __builtin_amdgcn_global_load_async_to_lds_b128(
      (v4i_vs*)gsrc, (v4i_vs*)ldst, /*imm offset*/ 0, /*cpol*/ 0);
}
#endif

// ---------------------------------------------------------------------------
// WMMA fragment loaders (CDNA5 ISA 7.12.2 layouts, wave32)
// ---------------------------------------------------------------------------
// A-matrix 16x32 bf16: lane half h, row m = lane&15.
//   VGPR j<4 : K = 2j + {0,1} + 8h ; VGPR j>=4 : K = 16 + 2(j-4) + {0,1} + 8h
__device__ __forceinline__ v16bf frag_a_load(const __bf16* base, int ld,
                                             int row0, int k0, int lane) {
  const int half = lane >> 4, m = lane & 15;
  const __bf16* p = base + (size_t)(row0 + m) * ld + k0;
  v16bf a;
#pragma unroll
  for (int j = 0; j < 8; ++j) {
    const int kb = ((j < 4) ? (2 * j) : (16 + 2 * (j - 4))) + 8 * half;
    a[2 * j]     = p[kb];
    a[2 * j + 1] = p[kb + 1];
  }
  return a;
}

// B-matrix 32x16 bf16 where B[k][n] = src[n0+n][k0+k] (src rows contiguous in k):
//   lane half h, col n = lane&15; element t<8: K = 8h + t ; t>=8: K = 16 + 8h + (t-8).
__device__ __forceinline__ v16bf frag_b_rows(const __bf16* base, int ld,
                                             int n0, int k0, int lane) {
  const int half = lane >> 4, n = lane & 15;
  const __bf16* p = base + (size_t)(n0 + n) * ld + k0 + 8 * half;
  v16bf b;
#pragma unroll
  for (int t = 0; t < 16; ++t) b[t] = p[(t < 8) ? t : (t + 8)];
  return b;
}

// ---------------------------------------------------------------------------
// f32 -> bf16 converter
// ---------------------------------------------------------------------------
__global__ __launch_bounds__(256) void cvt_f32_bf16(const float* __restrict__ in,
                                                    __bf16* __restrict__ out, int n) {
  for (int i = blockIdx.x * blockDim.x + threadIdx.x; i < n;
       i += gridDim.x * blockDim.x) {
    out[i] = (__bf16)in[i];
  }
}

// ---------------------------------------------------------------------------
// GEMM: C(M x 2048) = A(M x 2048, bf16 rowmajor) @ W(2048 x 2048, bf16 rowmajor)^T
// Block: 256 threads = 8 waves (2x4), 64x256 block tile, wave tile 32x64,
// K-step 32, double-buffered LDS filled by async global->LDS copies.
// Per K-step per wave: 2 A-frags + 4 B-frags -> 8 WMMAs.
// Epilogue modes:
//   0 (QK):  bf16 dst[(b*NH+h)*L + l][d]  (B,NH,L,DH)
//   1 (VT):  bf16 dst[(b*NH+h)*DH + d][l] (B,NH,DH,L)   (V transposed per head)
//   2 (F32): float dst[m][n]              (B*L, H)
// ---------------------------------------------------------------------------
#define GEMM_MODE_QK  0
#define GEMM_MODE_VT  1
#define GEMM_MODE_F32 2

__global__ __launch_bounds__(256) void gemm_xt(const __bf16* __restrict__ A,
                                               const __bf16* __restrict__ W,
                                               void* __restrict__ dstv, int mode) {
  constexpr int K = HH, N = HH;
  constexpr int LDT = 40;                 // padded LDS row stride (bf16 elems)
  constexpr int NT = K / 32;              // 64 K-tiles
  __shared__ __bf16 As[2][64 * LDT];      // 64 x 32 tile of A  (double buffered)
  __shared__ __bf16 Bs[2][256 * LDT];     // 256 x 32 tile of W (rows = out cols)

  const int t = threadIdx.x;
  const int lane = t & 31, wave = t >> 5;
  const int wr = wave >> 2, wc = wave & 3;      // 2 x 4 wave grid
  const int mBase = blockIdx.y * 64;
  const int nBase = blockIdx.x * 256;
  const int ar = t >> 2, aseg = t & 3;          // LDS fill indices

  // Issue the global->LDS copies for one 32-wide K tile into buffer `buf`.
  // 5 x b128 per thread: 1 for the A tile (64x32), 4 for the W tile (256x32).
  auto issue_tile = [&](int kt, int buf) {
    const __bf16* ga = A + (size_t)(mBase + ar) * K + kt + aseg * 8;
#if USE_ASYNC_LDS
    async_cp_b128(ga, &As[buf][ar * LDT + aseg * 8]);
#else
    *(uint4*)(&As[buf][ar * LDT + aseg * 8]) = *(const uint4*)ga;
#endif
#pragma unroll
    for (int i = 0; i < 4; ++i) {
      const int row = ar + 64 * i;
      const __bf16* gb = W + (size_t)(nBase + row) * K + kt + aseg * 8;
#if USE_ASYNC_LDS
      async_cp_b128(gb, &Bs[buf][row * LDT + aseg * 8]);
#else
      *(uint4*)(&Bs[buf][row * LDT + aseg * 8]) = *(const uint4*)gb;
#endif
    }
  };

  v8f acc[2][4] = {};

  issue_tile(0, 0);
  for (int it = 0; it < NT; ++it) {
    if (it + 1 < NT) {
      issue_tile((it + 1) * 32, (it + 1) & 1);
#if USE_ASYNC_LDS
      // 5 copies in flight for tile it+1; async loads complete in order per
      // wave, so <=5 outstanding means tile `it` has fully landed in LDS.
      __builtin_amdgcn_s_wait_asynccnt(5);
#endif
    } else {
#if USE_ASYNC_LDS
      __builtin_amdgcn_s_wait_asynccnt(0);
#endif
    }
    __syncthreads();   // all waves' copies for tile `it` visible

    const int buf = it & 1;
    const v16bf a0 = frag_a_load(As[buf], LDT, wr * 32, 0, lane);
    const v16bf a1 = frag_a_load(As[buf], LDT, wr * 32 + 16, 0, lane);
#pragma unroll
    for (int sj = 0; sj < 4; ++sj) {
      const v16bf b = frag_b_rows(Bs[buf], LDT, wc * 64 + 16 * sj, 0, lane);
      acc[0][sj] = WMMA_BF16(a0, b, acc[0][sj]);
      acc[1][sj] = WMMA_BF16(a1, b, acc[1][sj]);
    }
    __syncthreads();   // done reading buf before it is overwritten at it+2
  }

  const int half = lane >> 4, nl = lane & 15;
#pragma unroll
  for (int si = 0; si < 2; ++si)
#pragma unroll
    for (int sj = 0; sj < 4; ++sj)
#pragma unroll
      for (int r = 0; r < 8; ++r) {
        const int mg = mBase + wr * 32 + si * 16 + r + 8 * half;   // global row
        const int ng = nBase + wc * 64 + sj * 16 + nl;             // global col
        const float v = acc[si][sj][r];
        if (mode == GEMM_MODE_F32) {
          ((float*)dstv)[(size_t)mg * N + ng] = v;
        } else {
          const int b = mg >> 11, l = mg & (LL - 1);   // L = 2048
          const int h = ng >> 7,  d = ng & (DHH - 1);  // DH = 128
          __bf16* dst = (__bf16*)dstv;
          if (mode == GEMM_MODE_QK)
            dst[(((size_t)(b * NHH + h)) * LL + l) * DHH + d] = (__bf16)v;
          else
            dst[(((size_t)(b * NHH + h)) * DHH + d) * LL + l] = (__bf16)v;
        }
      }
}

// ---------------------------------------------------------------------------
// Flash attention: one wave per 16-query tile, key blocks of 32, online
// softmax, causal + padding mask, O accumulated in f32 via WMMA.
// Grid: (B*NH, L/64), block = 128 (4 waves).
// ---------------------------------------------------------------------------
__global__ __launch_bounds__(128) void flash_attn(const __bf16* __restrict__ Q,
                                                  const __bf16* __restrict__ Kb,
                                                  const __bf16* __restrict__ Vt,
                                                  const unsigned char* __restrict__ pmask,
                                                  __bf16* __restrict__ merged) {
  __shared__ __bf16 Pb[4][16 * 32];  // wave-private P staging (C-layout -> A-layout)

  const int lane = threadIdx.x & 31;
  const int wave = threadIdx.x >> 5;
  const int half = lane >> 4, nl = lane & 15;

  const int bh = blockIdx.x;   // 0..31 : b*NH + h
  const int b = bh >> 4;
  const int h = bh & 15;
  const __bf16* qh = Q  + (size_t)bh * LL * DHH;   // (L, DH)
  const __bf16* kh = Kb + (size_t)bh * LL * DHH;   // (L, DH)
  const __bf16* vh = Vt + (size_t)bh * DHH * LL;   // (DH, L)

  const int qt = blockIdx.y * 4 + wave;
  const int q0 = qt * 16;

  // Q fragments: 16 rows x 128 d, as 4 A-frags of K=32
  v16bf qf[4];
#pragma unroll
  for (int j = 0; j < 4; ++j) qf[j] = frag_a_load(qh, DHH, q0, 32 * j, lane);

  v8f o[8] = {};                 // 16 x 128 output accumulator (8 C-frags)
  float rm[8], rl[8];            // per-row running max / sum (row = r + 8*half)
#pragma unroll
  for (int r = 0; r < 8; ++r) { rm[r] = -__builtin_inff(); rl[r] = 0.f; }

  const float scale = 0.08838834764831845f;  // 1/sqrt(128)
  const int kbLast = (q0 + 15) >> 5;         // causal: keys <= q0+15

  for (int kb = 0; kb <= kbLast; ++kb) {
    const int key0 = kb * 32;

    if (kb < kbLast) {  // prefetch next key block of K (global_prefetch_b8)
      __builtin_prefetch(kh + (size_t)(key0 + 32 + lane) * DHH, 0, 3);
    }

    // ---- scores: S(16x32) = Q(16x128) @ K(32x128)^T -----------------------
    v8f s0 = {}, s1 = {};
#pragma unroll
    for (int j = 0; j < 4; ++j) {
      const v16bf bk0 = frag_b_rows(kh, DHH, key0,      32 * j, lane);
      s0 = WMMA_BF16(qf[j], bk0, s0);
      const v16bf bk1 = frag_b_rows(kh, DHH, key0 + 16, 32 * j, lane);
      s1 = WMMA_BF16(qf[j], bk1, s1);
    }

    // ---- scale + causal + padding masks -----------------------------------
    const int c0 = key0 + nl, c1 = key0 + 16 + nl;
    const bool pm0 = pmask[(size_t)b * LL + c0] != 0;
    const bool pm1 = pmask[(size_t)b * LL + c1] != 0;
#pragma unroll
    for (int r = 0; r < 8; ++r) {
      float f0 = s0[r] * scale, f1 = s1[r] * scale;
      const int row = q0 + r + 8 * half;
      if (pm0 || (kb == kbLast && c0 > row)) f0 = -__builtin_inff();
      if (pm1 || (kb == kbLast && c1 > row)) f1 = -__builtin_inff();
      s0[r] = f0; s1[r] = f1;
    }

    // ---- online softmax ---------------------------------------------------
    float alpha[8];
#pragma unroll
    for (int r = 0; r < 8; ++r) {
      float mx = fmaxf(s0[r], s1[r]);
#pragma unroll
      for (int off = 1; off < 16; off <<= 1) mx = fmaxf(mx, __shfl_xor(mx, off, 32));
      const float mnew = fmaxf(rm[r], mx);
      alpha[r] = (rm[r] == -__builtin_inff()) ? 0.f : __expf(rm[r] - mnew);
      const float p0 = (s0[r] == -__builtin_inff()) ? 0.f : __expf(s0[r] - mnew);
      const float p1 = (s1[r] == -__builtin_inff()) ? 0.f : __expf(s1[r] - mnew);
      float ps = p0 + p1;
#pragma unroll
      for (int off = 1; off < 16; off <<= 1) ps += __shfl_xor(ps, off, 32);
      rl[r] = rl[r] * alpha[r] + ps;
      rm[r] = mnew;
      // stash P (row m = r + 8*half, cols nl and 16+nl) for A-frag reload
      const int row_l = r + 8 * half;
      Pb[wave][row_l * 32 + nl]      = (__bf16)p0;
      Pb[wave][row_l * 32 + 16 + nl] = (__bf16)p1;
    }

    // rescale existing accumulator by alpha (per-row)
#pragma unroll
    for (int d = 0; d < 8; ++d)
#pragma unroll
      for (int r = 0; r < 8; ++r) o[d][r] *= alpha[r];

    // ---- O += P(16x32) @ V(32x128) ---------------------------------------
    const v16bf pf = frag_a_load(&Pb[wave][0], 32, 0, 0, lane);
#pragma unroll
    for (int d = 0; d < 8; ++d) {
      const v16bf bv = frag_b_rows(vh, LL, d * 16, key0, lane);  // B[k][d] = Vt[d][k]
      o[d] = WMMA_BF16(pf, bv, o[d]);
    }
  }

  // ---- normalize + store merged (B, L, H) bf16 ----------------------------
#pragma unroll
  for (int r = 0; r < 8; ++r) rl[r] = (rl[r] > 0.f) ? (1.f / rl[r]) : 0.f;
#pragma unroll
  for (int d = 0; d < 8; ++d)
#pragma unroll
    for (int r = 0; r < 8; ++r) {
      const int row = q0 + r + 8 * half;
      const int col = h * DHH + d * 16 + nl;
      merged[((size_t)b * LL + row) * HH + col] = (__bf16)(o[d][r] * rl[r]);
    }
}

// ---------------------------------------------------------------------------
// Host-side launch sequence (all on `stream`, graph-capture safe)
// ---------------------------------------------------------------------------
extern "C" void kernel_launch(void* const* d_in, const int* in_sizes, int n_in,
                              void* d_out, int out_size, void* d_ws, size_t ws_size,
                              hipStream_t stream) {
  (void)in_sizes; (void)n_in; (void)out_size; (void)ws_size;

  const float* x  = (const float*)d_in[0];
  const unsigned char* pmask = (const unsigned char*)d_in[1];
  const float* Wq = (const float*)d_in[2];
  const float* Wk = (const float*)d_in[3];
  const float* Wv = (const float*)d_in[4];
  const float* Wo = (const float*)d_in[5];
  float* out = (float*)d_out;

  // Workspace layout (bytes):
  //   [0,16M)   x_bf16  (4096x2048)  -- reused as `merged` after QKV GEMMs
  //   [16M,24M) w_bf16  (2048x2048)  -- reused for each weight sequentially
  //   [24M,40M) Q bf16  (B,NH,L,DH)
  //   [40M,56M) K bf16  (B,NH,L,DH)
  //   [56M,72M) Vt bf16 (B,NH,DH,L)
  char* ws = (char*)d_ws;
  __bf16* xbf = (__bf16*)(ws);
  __bf16* wbf = (__bf16*)(ws + (size_t)16 * 1024 * 1024);
  __bf16* qb  = (__bf16*)(ws + (size_t)24 * 1024 * 1024);
  __bf16* kb  = (__bf16*)(ws + (size_t)40 * 1024 * 1024);
  __bf16* vt  = (__bf16*)(ws + (size_t)56 * 1024 * 1024);
  __bf16* merged = xbf;  // alias: x no longer needed after QKV projections

  const int nX = MTOT * HH;   // 8388608
  const int nW = HH * HH;     // 4194304
  const dim3 gemmGrid(HH / 256, MTOT / 64);  // (8, 64)
  const dim3 attnGrid(NB * NHH, LL / 64);    // (32, 32)

  cvt_f32_bf16<<<2048, 256, 0, stream>>>(x, xbf, nX);

  cvt_f32_bf16<<<2048, 256, 0, stream>>>(Wq, wbf, nW);
  gemm_xt<<<gemmGrid, 256, 0, stream>>>(xbf, wbf, (void*)qb, GEMM_MODE_QK);

  cvt_f32_bf16<<<2048, 256, 0, stream>>>(Wk, wbf, nW);
  gemm_xt<<<gemmGrid, 256, 0, stream>>>(xbf, wbf, (void*)kb, GEMM_MODE_QK);

  cvt_f32_bf16<<<2048, 256, 0, stream>>>(Wv, wbf, nW);
  gemm_xt<<<gemmGrid, 256, 0, stream>>>(xbf, wbf, (void*)vt, GEMM_MODE_VT);

  flash_attn<<<attnGrid, 128, 0, stream>>>(qb, kb, vt, pmask, merged);

  cvt_f32_bf16<<<2048, 256, 0, stream>>>(Wo, wbf, nW);
  gemm_xt<<<gemmGrid, 256, 0, stream>>>(merged, wbf, (void*)out, GEMM_MODE_F32);
}
